// MultiHeadSelfAttention_35777077576207
// MI455X (gfx1250) — 8-bit matrix operand order repaired and checked by reading, compiled, not run on hardware
//
#include <hip/hip_runtime.h>
#include <hip/hip_bf16.h>
#include <stdint.h>

typedef int v8i __attribute__((ext_vector_type(8)));

#define NB    16
#define SEQ   512
#define HID   1024
#define NHEAD 16
#define DHEAD 64
#define MROW  (NB*SEQ)
#define NSEQT 16


__device__ __forceinline__ float scale_from_abs(unsigned bits) {
    float s = __uint_as_float(bits) * (1.0f / 127.0f);
    return (s == 0.0f) ? 1.0f : s;
}
__device__ __forceinline__ float warp_max(float v) {
    for (int o = 16; o; o >>= 1) v = fmaxf(v, __shfl_xor(v, o, 32));
    return v;
}
__device__ __forceinline__ float warp_sum(float v) {
    for (int o = 16; o; o >>= 1) v += __shfl_xor(v, o, 32);
    return v;
}
__device__ __forceinline__ float block_max256(float v) {
    __shared__ float sm[8];
    v = warp_max(v);
    if ((threadIdx.x & 31) == 0) sm[threadIdx.x >> 5] = v;
    __syncthreads();
    float r = (threadIdx.x < 8) ? sm[threadIdx.x] : -3.4e38f;
    r = warp_max(r);
    return r;
}

__global__ void k_init(unsigned* scal, float* pla) {
    int t = threadIdx.x;
    if (t < 64) scal[t] = 0u;
    __syncthreads();
    if (t == 0) scal[3] = 0x7f7fffffu;
    if (t < 12) {
        double e0 = -10.0 + t * (10.0 / 12.0);
        double e1 = -10.0 + (t + 1) * (10.0 / 12.0);
        double n = 0, sx = 0, sy = 0, sxx = 0, sxy = 0;
        for (int i = 0; i <= 1000; i++) {
            double x = -10.0 + 0.01 * i;
            if (x >= e0 && x <= e1) {
                double y = exp(x);
                n += 1.0; sx += x; sy += y; sxx += x * x; sxy += x * y;
            }
        }
        double m = (n * sxy - sx * sy) / (n * sxx - sx * sx);
        double c = (sy - m * sx) / n;
        pla[t] = (float)m;
        pla[12 + t] = (float)c;
    }
}

__global__ void k_absmax(const float* __restrict__ x, long n, unsigned* slot) {
    float v = 0.f;
    long stride = (long)gridDim.x * blockDim.x;
    for (long i = (long)blockIdx.x * blockDim.x + threadIdx.x; i < n; i += stride)
        v = fmaxf(v, fabsf(x[i]));
    v = block_max256(v);
    if (threadIdx.x == 0) atomicMax(slot, __float_as_uint(v));
}

__global__ void k_quant(const float* __restrict__ x, long n,
                        const unsigned* __restrict__ slot, int8_t* __restrict__ out) {
    float inv = 1.0f / scale_from_abs(*slot);
    long stride = (long)gridDim.x * blockDim.x;
    for (long i = (long)blockIdx.x * blockDim.x + threadIdx.x; i < n; i += stride) {
        float q = rintf(x[i] * inv);
        q = fminf(127.f, fmaxf(-127.f, q));
        out[i] = (int8_t)q;
    }
}

__global__ void k_rowquant(const float* __restrict__ W, int8_t* __restrict__ Wq,
                           float* __restrict__ rsc) {
    int row = blockIdx.x;
    const float* wr = W + (size_t)row * HID;
    float v = 0.f;
    for (int k = threadIdx.x; k < HID; k += 256) v = fmaxf(v, fabsf(wr[k]));
    v = block_max256(v);
    __shared__ float ssc;
    if (threadIdx.x == 0) {
        float s = v * (1.0f / 127.0f);
        if (s == 0.f) s = 1.f;
        rsc[row] = s;
        ssc = s;
    }
    __syncthreads();
    float inv = 1.0f / ssc;
    for (int k = threadIdx.x; k < HID; k += 256) {
        float q = rintf(wr[k] * inv);
        q = fminf(127.f, fmaxf(-127.f, q));
        Wq[(size_t)row * HID + k] = (int8_t)q;
    }
}

__global__ __launch_bounds__(256) void k_gemm_iu8(
    const int8_t* __restrict__ A, const int8_t* __restrict__ Bw,
    const unsigned* __restrict__ aAbs, const float* __restrict__ bScale,
    const float* __restrict__ bias, float* __restrict__ out, int bhsd) {
    const int lane = threadIdx.x & 31, wave = threadIdx.x >> 5;
    const int g = lane >> 4, ln = lane & 15;
    const int m0 = blockIdx.y * 128 + wave * 16;
    const int n0 = blockIdx.x * 64;
    const int8_t* arow = A + (size_t)(m0 + ln) * HID + g * 16;
    v8i zero = {0, 0, 0, 0, 0, 0, 0, 0};
    v8i acc[4];
    acc[0] = zero; acc[1] = zero; acc[2] = zero; acc[3] = zero;
    for (int ks = 0; ks < HID / 64; ks++) {
        const int8_t* ap = arow + ks * 64;
        __builtin_prefetch(ap + 64, 0, 1);
        uint4 a0 = *(const uint4*)(ap);
        uint4 a1 = *(const uint4*)(ap + 32);
        v8i av;
        av[0] = (int)a0.x; av[1] = (int)a0.y; av[2] = (int)a0.z; av[3] = (int)a0.w;
        av[4] = (int)a1.x; av[5] = (int)a1.y; av[6] = (int)a1.z; av[7] = (int)a1.w;
#pragma unroll
        for (int nt = 0; nt < 4; nt++) {
            const int8_t* bp = Bw + (size_t)(n0 + nt * 16 + ln) * HID + ks * 64 + g * 16;
            uint4 b0 = *(const uint4*)(bp);
            uint4 b1 = *(const uint4*)(bp + 32);
            v8i bv;
            bv[0] = (int)b0.x; bv[1] = (int)b0.y; bv[2] = (int)b0.z; bv[3] = (int)b0.w;
            bv[4] = (int)b1.x; bv[5] = (int)b1.y; bv[6] = (int)b1.z; bv[7] = (int)b1.w;
            acc[nt] = __builtin_amdgcn_wmma_i32_16x16x64_iu8(true, av, true, bv, acc[nt],
                                                             false, false);
        }
    }
    float as = scale_from_abs(*aAbs);
#pragma unroll
    for (int nt = 0; nt < 4; nt++) {
        int col = n0 + nt * 16 + ln;
        float sc = as * bScale[col];
        float bb = bias[col];
#pragma unroll
        for (int v = 0; v < 8; v++) {
            int row = m0 + v + 8 * g;
            float f = (float)acc[nt][v] * sc + bb;
            if (bhsd) {
                int b = row >> 9, s = row & 511, h = col >> 6, dd = col & 63;
                out[((((size_t)b * NHEAD + h) * SEQ) + s) * DHEAD + dd] = f;
            } else {
                out[(size_t)row * HID + col] = f;
            }
        }
    }
}

__global__ void k_tileabs(const float* __restrict__ x, unsigned* __restrict__ slots) {
    long base = (long)blockIdx.x * 2048;
    int tile = blockIdx.x & 15;
    float v = 0.f;
    for (int i = threadIdx.x; i < 2048; i += 256) v = fmaxf(v, fabsf(x[base + i]));
    v = block_max256(v);
    if (threadIdx.x == 0) atomicMax(&slots[tile], __float_as_uint(v));
}

__global__ void k_quant_qk(const float* __restrict__ x, const unsigned* __restrict__ slots,
                           int8_t* __restrict__ out) {
    long n = (long)MROW * HID;
    long stride = (long)gridDim.x * blockDim.x;
    for (long i = (long)blockIdx.x * blockDim.x + threadIdx.x; i < n; i += stride) {
        int tile = (int)((i >> 11) & 15);
        float inv = 1.0f / scale_from_abs(slots[tile]);
        float q = rintf(x[i] * inv);
        q = fminf(127.f, fmaxf(-127.f, q));
        out[i] = (int8_t)q;
    }
}

__global__ void k_quant_vt(const float* __restrict__ vf, const unsigned* __restrict__ slot,
                           int8_t* __restrict__ vt) {
    float inv = 1.0f / scale_from_abs(*slot);
    long n = (long)MROW * HID;
    long stride = (long)gridDim.x * blockDim.x;
    for (long o = (long)blockIdx.x * blockDim.x + threadIdx.x; o < n; o += stride) {
        long s = o & 511;
        long d = (o >> 9) & 63;
        long bh = o >> 15;
        float x = vf[(bh << 15) + s * 64 + d];
        float q = rintf(x * inv);
        q = fminf(127.f, fmaxf(-127.f, q));
        vt[o] = (int8_t)q;
    }
}

__global__ void k_finalize(unsigned* scal, const float* pla) {
    if (threadIdx.x == 0) {
        float ms = __uint_as_float(scal[3]);
        float mxp = pla[23] / (ms + 1e-9f);
        float ps = mxp * (1.0f / 127.0f);
        if (!(ps > 0.f)) ps = 1.f;
        scal[4] = __float_as_uint(ps);
    }
}

__device__ __forceinline__ unsigned pack4q(float a, float b, float c, float d, float s) {
    int q0 = (int)fminf(127.f, fmaxf(-127.f, rintf(a * s)));
    int q1 = (int)fminf(127.f, fmaxf(-127.f, rintf(b * s)));
    int q2 = (int)fminf(127.f, fmaxf(-127.f, rintf(c * s)));
    int q3 = (int)fminf(127.f, fmaxf(-127.f, rintf(d * s)));
    return (unsigned)(q0 & 255) | ((unsigned)(q1 & 255) << 8) |
           ((unsigned)(q2 & 255) << 16) | ((unsigned)(q3 & 255) << 24);
}

__global__ __launch_bounds__(32) void k_attn(
    const int8_t* __restrict__ Qi, const int8_t* __restrict__ Ki,
    const int8_t* __restrict__ VT, const unsigned* __restrict__ scal,
    const float* __restrict__ pla, unsigned* __restrict__ minslot,
    float* __restrict__ ctx, int pass) {
    __shared__ float lds[16][512];
    __shared__ float srow[16];
    const int lane = threadIdx.x;
    const int qt = blockIdx.x, h = blockIdx.y, b = blockIdx.z;
    const int bh = b * NHEAD + h;
    const int g = lane >> 4, ln = lane & 15;
    const int qbase = qt * 16;

    const int8_t* ap = Qi + ((size_t)bh * SEQ + qbase + ln) * DHEAD + g * 16;
    uint4 a0 = *(const uint4*)(ap);
    uint4 a1 = *(const uint4*)(ap + 32);
    v8i av;
    av[0] = (int)a0.x; av[1] = (int)a0.y; av[2] = (int)a0.z; av[3] = (int)a0.w;
    av[4] = (int)a1.x; av[5] = (int)a1.y; av[6] = (int)a1.z; av[7] = (int)a1.w;

    const float sq = scale_from_abs(scal[5 + (qt >> 1)]);
    v8i zero = {0, 0, 0, 0, 0, 0, 0, 0};

    for (int nt = 0; nt < 32; nt++) {
        const int8_t* bp = Ki + ((size_t)bh * SEQ + nt * 16 + ln) * DHEAD + g * 16;
        uint4 b0 = *(const uint4*)(bp);
        uint4 b1 = *(const uint4*)(bp + 32);
        v8i bv;
        bv[0] = (int)b0.x; bv[1] = (int)b0.y; bv[2] = (int)b0.z; bv[3] = (int)b0.w;
        bv[4] = (int)b1.x; bv[5] = (int)b1.y; bv[6] = (int)b1.z; bv[7] = (int)b1.w;
        v8i c = __builtin_amdgcn_wmma_i32_16x16x64_iu8(true, av, true, bv, zero, false, false);
        float sk = scale_from_abs(scal[21 + (nt >> 1)]);
        float sc = sq * sk * 0.125f;
#pragma unroll
        for (int v = 0; v < 8; v++) lds[v + 8 * g][nt * 16 + ln] = (float)c[v] * sc;
    }
    __syncthreads();

    const float* plm = pla;
    const float* plc = pla + 12;
    for (int r = 0; r < 16; r++) {
        float mx = -3.4e38f;
        for (int j = 0; j < 16; j++) mx = fmaxf(mx, lds[r][lane + 32 * j]);
        mx = warp_max(mx);
        float sum = 0.f;
        for (int j = 0; j < 16; j++) {
            int i = lane + 32 * j;
            float xc = lds[r][i] - mx;
            xc = fminf(0.f, fmaxf(-10.f, xc));
            int ii = (int)((xc + 10.f) * 1.2f);
            ii = ii > 11 ? 11 : ii;
            float e = plm[ii] * xc + plc[ii];
            lds[r][i] = e;
            sum += e;
        }
        sum = warp_sum(sum);
        if (lane == 0) {
            srow[r] = sum;
            if (pass == 1) atomicMin(minslot, __float_as_uint(sum));
        }
    }
    __syncthreads();
    if (pass == 1) return;

    const float ps = __uint_as_float(scal[4]);
    const float qs = (1.0f / ps) * (1.0f / (srow[ln] + 1e-9f));
    v8i acc[4];
    acc[0] = zero; acc[1] = zero; acc[2] = zero; acc[3] = zero;
    for (int kk = 0; kk < 8; kk++) {
        v8i av2;
#pragma unroll
        for (int c4 = 0; c4 < 4; c4++) {
            int kb = kk * 64 + (c4 >> 1) * 32 + g * 16 + (c4 & 1) * 8;
            float4 f0 = *(const float4*)&lds[ln][kb];
            float4 f1 = *(const float4*)&lds[ln][kb + 4];
            av2[2 * c4] = (int)pack4q(f0.x, f0.y, f0.z, f0.w, qs);
            av2[2 * c4 + 1] = (int)pack4q(f1.x, f1.y, f1.z, f1.w, qs);
        }
#pragma unroll
        for (int dt = 0; dt < 4; dt++) {
            const int8_t* vp = VT + ((size_t)bh * DHEAD + dt * 16 + ln) * SEQ + kk * 64 + g * 16;
            uint4 b0 = *(const uint4*)(vp);
            uint4 b1 = *(const uint4*)(vp + 32);
            v8i bv;
            bv[0] = (int)b0.x; bv[1] = (int)b0.y; bv[2] = (int)b0.z; bv[3] = (int)b0.w;
            bv[4] = (int)b1.x; bv[5] = (int)b1.y; bv[6] = (int)b1.z; bv[7] = (int)b1.w;
            acc[dt] = __builtin_amdgcn_wmma_i32_16x16x64_iu8(true, av2, true, bv, acc[dt],
                                                             false, false);
        }
    }
    const float sv = scale_from_abs(scal[1]);
    const float os = ps * sv;
#pragma unroll
    for (int dt = 0; dt < 4; dt++) {
        int col = h * DHEAD + dt * 16 + ln;
#pragma unroll
        for (int v = 0; v < 8; v++) {
            int s = qbase + v + 8 * g;
            ctx[(((size_t)b * SEQ) + s) * HID + col] = (float)acc[dt][v] * os;
        }
    }
}

extern "C" void kernel_launch(void* const* d_in, const int* in_sizes, int n_in,
                              void* d_out, int out_size, void* d_ws, size_t ws_size,
                              hipStream_t stream) {
    (void)in_sizes; (void)n_in; (void)out_size; (void)ws_size;
    const float* hs = (const float*)d_in[0];
    const float* Wq = (const float*)d_in[1]; const float* bq = (const float*)d_in[2];
    const float* Wk = (const float*)d_in[3]; const float* bk = (const float*)d_in[4];
    const float* Wv = (const float*)d_in[5]; const float* bv = (const float*)d_in[6];
    const float* Wo = (const float*)d_in[7]; const float* bo = (const float*)d_in[8];
    float* out = (float*)d_out;

    char* base = (char*)d_ws;
    size_t off = 0;
    auto carve = [&](size_t bytes) -> char* {
        char* p = base + off;
        off = (off + bytes + 255) & ~(size_t)255;
        return p;
    };
    const long nAct = (long)MROW * HID;
    unsigned* scal = (unsigned*)carve(64 * 4);
    float* pla = (float*)carve(48 * 4);
    float* wsc = (float*)carve(4 * HID * 4);
    int8_t* Xq  = (int8_t*)carve((size_t)nAct);
    int8_t* Wqi = (int8_t*)carve((size_t)HID * HID);
    int8_t* Wki = (int8_t*)carve((size_t)HID * HID);
    int8_t* Wvi = (int8_t*)carve((size_t)HID * HID);
    int8_t* Woi = (int8_t*)carve((size_t)HID * HID);
    float* QF = (float*)carve((size_t)nAct * 4);
    float* KF = (float*)carve((size_t)nAct * 4);
    float* VF = (float*)carve((size_t)nAct * 4);
    int8_t* Qi8 = (int8_t*)carve((size_t)nAct);
    int8_t* Ki8 = (int8_t*)carve((size_t)nAct);
    int8_t* VT8 = (int8_t*)carve((size_t)nAct);
    int8_t* Cq  = (int8_t*)carve((size_t)nAct);

    k_init<<<1, 64, 0, stream>>>(scal, pla);
    k_absmax<<<1024, 256, 0, stream>>>(hs, nAct, scal + 0);
    k_quant<<<1024, 256, 0, stream>>>(hs, nAct, scal + 0, Xq);
    k_rowquant<<<HID, 256, 0, stream>>>(Wq, Wqi, wsc + 0);
    k_rowquant<<<HID, 256, 0, stream>>>(Wk, Wki, wsc + HID);
    k_rowquant<<<HID, 256, 0, stream>>>(Wv, Wvi, wsc + 2 * HID);
    k_rowquant<<<HID, 256, 0, stream>>>(Wo, Woi, wsc + 3 * HID);

    dim3 gg(HID / 64, MROW / 128);
    k_gemm_iu8<<<gg, 256, 0, stream>>>(Xq, Wqi, scal + 0, wsc + 0, bq, QF, 1);
    k_gemm_iu8<<<gg, 256, 0, stream>>>(Xq, Wki, scal + 0, wsc + HID, bk, KF, 1);
    k_gemm_iu8<<<gg, 256, 0, stream>>>(Xq, Wvi, scal + 0, wsc + 2 * HID, bv, VF, 1);

    k_tileabs<<<4096, 256, 0, stream>>>(QF, scal + 5);
    k_tileabs<<<4096, 256, 0, stream>>>(KF, scal + 21);
    k_absmax<<<1024, 256, 0, stream>>>(VF, nAct, scal + 1);
    k_quant_qk<<<2048, 256, 0, stream>>>(QF, scal + 5, Qi8);
    k_quant_qk<<<2048, 256, 0, stream>>>(KF, scal + 21, Ki8);
    k_quant_vt<<<2048, 256, 0, stream>>>(VF, scal + 1, VT8);

    dim3 ga(SEQ / 16, NHEAD, NB);
    k_attn<<<ga, 32, 0, stream>>>(Qi8, Ki8, VT8, scal, pla, scal + 3, QF, 1);
    k_finalize<<<1, 32, 0, stream>>>(scal, pla);
    k_attn<<<ga, 32, 0, stream>>>(Qi8, Ki8, VT8, scal, pla, scal + 3, QF, 2);

    k_absmax<<<1024, 256, 0, stream>>>(QF, nAct, scal + 2);
    k_quant<<<1024, 256, 0, stream>>>(QF, nAct, scal + 2, Cq);
    k_gemm_iu8<<<gg, 256, 0, stream>>>(Cq, Woi, scal + 2, wsc + 3 * HID, bo, out, 0);
}
